// CausalAttNet_51685636440683
// MI455X (gfx1250) — matrix operand order repaired and checked by reading, not compiled, not run on hardware
//
#include <hip/hip_runtime.h>
#include <hip/hip_bf16.h>

// ---------------- problem constants ----------------
#define NG      1024          // graphs
#define NPG     64            // nodes per graph
#define EPG     256           // edges per graph
#define CD      128           // channels
#define IND     32            // input dim
#define NEMB    8             // edge-attr table size
#define KSEL    128           // top-K edges kept per graph
#define NN      (NG * NPG)    // 65536 nodes
#define NEDGE   (NG * EPG)    // 262144 edges

// output layout (floats)
#define HN        (NN * CD)                 // 8388608
#define OFF_C_H   0L
#define OFF_C_EI  (OFF_C_H  + HN)           // 2*G*K = 262144
#define OFF_C_EA  (OFF_C_EI + 2L*NG*KSEL)   // 131072
#define OFF_C_W   (OFF_C_EA + (long)NG*KSEL)
#define OFF_C_B   (OFF_C_W  + (long)NG*KSEL)
#define OFF_F_H   (OFF_C_B  + (long)NN)
#define OFF_F_EI  (OFF_F_H  + HN)
#define OFF_F_EA  (OFF_F_EI + 2L*NG*KSEL)
#define OFF_F_W   (OFF_F_EA + (long)NG*KSEL)
#define OFF_F_B   (OFF_F_W  + (long)NG*KSEL)
#define OFF_W     (OFF_F_B  + (long)NN)

// ---------------- WMMA types ----------------
typedef __attribute__((ext_vector_type(16))) __bf16 v16bf;
typedef __attribute__((ext_vector_type(8)))  __bf16 v8bf;
typedef __attribute__((ext_vector_type(8)))  float  v8f;

union AFrag { v16bf v; v8bf h[2]; };

__device__ __forceinline__ v8f wmma_bf16(const v16bf& a, const v16bf& b, v8f c) {
  return __builtin_amdgcn_wmma_f32_16x16x32_bf16(false, a, false, b, (short)0, c,
                                                 false, false);
}

// ---------------- weight conversion: f32 [K][N] row-major -> bf16 [N][K] col-major
__global__ void wconvert_kernel(const float* __restrict__ W, __bf16* __restrict__ out,
                                int K, int N) {
  int i = blockIdx.x * 256 + threadIdx.x;
  if (i < K * N) {
    int n = i / K, k = i - n * K;
    out[n * K + k] = (__bf16)W[k * N + n];
  }
}

// ---------------- encoder: h = x @ W_enc + b_enc  (K=32, one WMMA per tile)
__global__ void __launch_bounds__(128)
encode_kernel(const float* __restrict__ x, const __bf16* __restrict__ Wcm,
              const float* __restrict__ benc, float* __restrict__ h) {
  __shared__ __bf16 A[64 * IND];
  const int tid = threadIdx.x;
  const int base = blockIdx.x * 64;                 // row stripe
  for (int i = tid; i < 64 * IND; i += 128)
    A[i] = (__bf16)x[(long)base * IND + i];
  __syncthreads();

  const int wave = tid >> 5, lane = tid & 31;
  const int hi = lane >> 4, lr = lane & 15;
  const int m0 = wave * 16;
  AFrag a;
  const int r = m0 + lr;
  a.v = *(const v16bf*)&A[r * IND + hi * 16];

  for (int n0 = 0; n0 < CD; n0 += 16) {
    const int n = n0 + lr;
    v16bf b = *(const v16bf*)&Wcm[n * IND + hi * 16];
    v8f acc = {};
    acc = wmma_bf16(a.v, b, acc);
    const float bias = benc[n];
    const int mrow = base + m0 + hi * 8;
#pragma unroll
    for (int rr = 0; rr < 8; ++rr)
      h[(long)(mrow + rr) * CD + n] = acc[rr] + bias;
  }
}

// ---------------- h += vn[batch]
__global__ void addvn_kernel(float* __restrict__ h, const float* __restrict__ vn) {
  long i = (long)blockIdx.x * 256 + threadIdx.x;    // over float4 elems
  if (i < (long)NN * CD / 4) {
    long n = i >> 5;           // node
    int  j = (int)(i & 31);    // float4 lane in channel dim
    const float4* v4 = (const float4*)vn;
    float4* h4 = (float4*)h;
    float4 a = h4[i], b = v4[(n >> 6) * 32 + j];
    a.x += b.x; a.y += b.y; a.z += b.z; a.w += b.w;
    h4[i] = a;
  }
}

// ---------------- message + aggregate (one block per graph, LDS accumulator)
__global__ void __launch_bounds__(256)
message_agg_kernel(const float* __restrict__ h, const int* __restrict__ row,
                   const int* __restrict__ col, const int* __restrict__ attr,
                   const float* __restrict__ tab, float* __restrict__ aggout) {
  __shared__ float agg[NPG * CD];                   // 32 KB
  const int tid = threadIdx.x, g = blockIdx.x;
  for (int i = tid; i < NPG * CD; i += 256) agg[i] = 0.f;
  __syncthreads();

  const int lane = tid & 31, eslot = tid >> 5;      // 8 edges in flight
  const float4* h4 = (const float4*)h;
  const float4* t4 = (const float4*)tab;
  for (int it = 0; it < EPG / 8; ++it) {
    const int e = g * EPG + it * 8 + eslot;
    const int c  = col[e];
    const int rl = row[e] - g * NPG;
    const int a  = attr[e];
    float4 hv = h4[(long)c * (CD / 4) + lane];
    float4 tv = t4[a * (CD / 4) + lane];
    float m0 = fmaxf(hv.x + tv.x, 0.f);
    float m1 = fmaxf(hv.y + tv.y, 0.f);
    float m2 = fmaxf(hv.z + tv.z, 0.f);
    float m3 = fmaxf(hv.w + tv.w, 0.f);
    float* dst = &agg[rl * CD + lane * 4];
    atomicAdd(dst + 0, m0);                         // ds_add_f32
    atomicAdd(dst + 1, m1);
    atomicAdd(dst + 2, m2);
    atomicAdd(dst + 3, m3);
  }
  __syncthreads();
  float4* o4 = (float4*)(aggout + (long)g * NPG * CD);
  const float4* a4 = (const float4*)agg;
  for (int i = tid; i < NPG * CD / 4; i += 256) o4[i] = a4[i];
}

// ---------------- fused MLP: hout = [relu]( relu((h+agg)@W1+b1) @ W2 + b2 )
__global__ void __launch_bounds__(128)
mlp_kernel(const float* __restrict__ h, const float* __restrict__ agg,
           const __bf16* __restrict__ W1cm, const float* __restrict__ b1,
           const __bf16* __restrict__ W2cm, const float* __restrict__ b2,
           float* __restrict__ hout, int relu_out) {
  __shared__ __bf16 A[64 * CD];                     // 16 KB
  __shared__ __bf16 Z[64 * 2 * CD];                 // 32 KB
  const int tid = threadIdx.x;
  const long base = (long)blockIdx.x * 64 * CD;
  __builtin_prefetch(W1cm, 0, 1);
  __builtin_prefetch(W2cm, 0, 1);
  for (int i = tid; i < 64 * CD; i += 128)
    A[i] = (__bf16)(h[base + i] + agg[base + i]);
  __syncthreads();

  const int wave = tid >> 5, lane = tid & 31;
  const int hi = lane >> 4, lr = lane & 15;
  const int m0 = wave * 16;
  const int r = m0 + lr;

  // ---- GEMM1: (64x128) x (128x256) -> relu -> Z (bf16)
  AFrag af[4];
#pragma unroll
  for (int kc = 0; kc < 4; ++kc) {
    af[kc].v = *(const v16bf*)&A[r * CD + kc * 32 + hi * 16];
  }
  for (int n0 = 0; n0 < 2 * CD; n0 += 16) {
    const int n = n0 + lr;
    v8f acc = {};
#pragma unroll
    for (int kc = 0; kc < 4; ++kc) {
      v16bf b = *(const v16bf*)&W1cm[n * CD + kc * 32 + hi * 16];
      acc = wmma_bf16(af[kc].v, b, acc);
    }
    const float bias = b1[n];
#pragma unroll
    for (int rr = 0; rr < 8; ++rr) {
      float z = acc[rr] + bias;
      Z[(m0 + hi * 8 + rr) * (2 * CD) + n] = (__bf16)(z > 0.f ? z : 0.f);
    }
  }
  __syncthreads();

  // ---- GEMM2: (64x256) x (256x128) -> hout
  AFrag zf[8];
#pragma unroll
  for (int kc = 0; kc < 8; ++kc) {
    zf[kc].v = *(const v16bf*)&Z[r * (2 * CD) + kc * 32 + hi * 16];
  }
  for (int n0 = 0; n0 < CD; n0 += 16) {
    const int n = n0 + lr;
    v8f acc = {};
#pragma unroll
    for (int kc = 0; kc < 8; ++kc) {
      v16bf b = *(const v16bf*)&W2cm[n * (2 * CD) + kc * 32 + hi * 16];
      acc = wmma_bf16(zf[kc].v, b, acc);
    }
    const float bias = b2[n];
#pragma unroll
    for (int rr = 0; rr < 8; ++rr) {
      float v = acc[rr] + bias;
      if (relu_out) v = v > 0.f ? v : 0.f;
      hout[base + (m0 + hi * 8 + rr) * CD + n] = v;
    }
  }
}

// ---------------- virtual node: vn = relu(segment_sum(h, batch) @ Wvn + bvn)
__global__ void __launch_bounds__(128)
vnode_kernel(const float* __restrict__ h, const float* __restrict__ Wvn,
             const float* __restrict__ bvn, float* __restrict__ vn) {
  __shared__ float pooled[CD];
  const int c = threadIdx.x, g = blockIdx.x;
  float s = 0.f;
  const float* hb = h + (long)g * NPG * CD;
  for (int i = 0; i < NPG; ++i) s += hb[i * CD + c];
  pooled[c] = s;
  __syncthreads();
  float acc = bvn[c];
  for (int k = 0; k < CD; ++k) acc += pooled[k] * Wvn[k * CD + c];
  vn[(long)g * CD + c] = acc > 0.f ? acc : 0.f;
}

// ---------------- per-node score halves: s0 = h.Wl[0:128], s1 = h.Wl[128:256]
__global__ void __launch_bounds__(256)
score_kernel(const float* __restrict__ h, const float* __restrict__ Wl,
             float* __restrict__ s0, float* __restrict__ s1) {
  __shared__ float wl[2 * CD];
  const int tid = threadIdx.x;
  if (tid < 2 * CD) wl[tid] = Wl[tid];
  __syncthreads();
  const int n = blockIdx.x * 256 + tid;
  if (n < NN) {
    const float4* h4 = (const float4*)(h + (long)n * CD);
    float a0 = 0.f, a1 = 0.f;
#pragma unroll 4
    for (int j = 0; j < CD / 4; ++j) {
      float4 v = h4[j];
      a0 += v.x * wl[j * 4 + 0] + v.y * wl[j * 4 + 1] +
            v.z * wl[j * 4 + 2] + v.w * wl[j * 4 + 3];
      a1 += v.x * wl[CD + j * 4 + 0] + v.y * wl[CD + j * 4 + 1] +
            v.z * wl[CD + j * 4 + 2] + v.w * wl[CD + j * 4 + 3];
    }
    s0[n] = a0;
    s1[n] = a1;
  }
}

// ---------------- edge scores w[e] = s0[row] + s1[col] + b
__global__ void edgew_kernel(const int* __restrict__ row, const int* __restrict__ col,
                             const float* __restrict__ s0, const float* __restrict__ s1,
                             const float* __restrict__ blin, float* __restrict__ wv,
                             float* __restrict__ out_w) {
  int e = blockIdx.x * 256 + threadIdx.x;
  if (e < NEDGE) {
    float w = s0[row[e]] + s1[col[e]] + blin[0];
    wv[e] = w;
    out_w[e] = w;                                   // "w" output (third tuple member)
  }
}

// ---------------- per-graph bitonic top-K + selection writer
__global__ void __launch_bounds__(256)
select_kernel(const float* __restrict__ wv, const int* __restrict__ row,
              const int* __restrict__ col, const int* __restrict__ attr,
              float* __restrict__ out) {
  __shared__ float val[EPG];
  __shared__ int   idx[EPG];
  const int t = threadIdx.x, g = blockIdx.x;
  val[t] = wv[(long)g * EPG + t];
  idx[t] = t;
  __syncthreads();
  // bitonic sort, final order: descending value, ties -> ascending index
  for (int k = 2; k <= EPG; k <<= 1) {
    for (int j = k >> 1; j > 0; j >>= 1) {
      int p = t ^ j;
      if (p > t) {
        bool desc = ((t & k) == 0);
        float v0 = val[t], v1 = val[p];
        int i0 = idx[t], i1 = idx[p];
        bool after = (v0 < v1) || (v0 == v1 && i0 > i1);   // v0 belongs after v1
        if (after == desc) { val[t] = v1; val[p] = v0; idx[t] = i1; idx[p] = i0; }
      }
      __syncthreads();
    }
  }
  const int es = idx[t];
  const long eid = (long)g * EPG + es;
  const float rw = (float)row[eid], cw = (float)col[eid], aw = (float)attr[eid];
  if (t < KSEL) {                                   // causal (kept) edges
    const long pos = (long)g * KSEL + t;
    out[OFF_C_EI + pos] = rw;
    out[OFF_C_EI + (long)NG * KSEL + pos] = cw;
    out[OFF_C_EA + pos] = aw;
    out[OFF_C_W + pos] = val[t];
  } else {                                          // conf (dropped) edges
    const long pos = (long)g * KSEL + (t - KSEL);
    out[OFF_F_EI + pos] = rw;
    out[OFF_F_EI + (long)NG * KSEL + pos] = cw;
    out[OFF_F_EA + pos] = aw;
    out[OFF_F_W + pos] = -val[t];
  }
}

// ---------------- output copies: h twice, batch twice
__global__ void copyh_kernel(const float* __restrict__ h, float* __restrict__ out) {
  long i = (long)blockIdx.x * 256 + threadIdx.x;    // over float4
  if (i < (long)HN / 4) {
    float4 v = ((const float4*)h)[i];
    ((float4*)(out + OFF_C_H))[i] = v;
    ((float4*)(out + OFF_F_H))[i] = v;
  }
}
__global__ void batch_kernel(float* __restrict__ out) {
  int n = blockIdx.x * 256 + threadIdx.x;
  if (n < NN) {
    float b = (float)(n >> 6);
    out[OFF_C_B + n] = b;
    out[OFF_F_B + n] = b;
  }
}

// ---------------- host launcher ----------------
extern "C" void kernel_launch(void* const* d_in, const int* in_sizes, int n_in,
                              void* d_out, int out_size, void* d_ws, size_t ws_size,
                              hipStream_t stream) {
  const float* x     = (const float*)d_in[0];
  const int*   eidx  = (const int*)d_in[1];
  const int*   row   = eidx;
  const int*   col   = eidx + NEDGE;
  const int*   attr  = (const int*)d_in[2];
  const float* Wenc  = (const float*)d_in[4];
  const float* benc  = (const float*)d_in[5];
  const float* tab0  = (const float*)d_in[6];
  const float* tab1  = (const float*)d_in[7];
  const float* W1_0  = (const float*)d_in[8];
  const float* b1_0  = (const float*)d_in[9];
  const float* W2_0  = (const float*)d_in[10];
  const float* b2_0  = (const float*)d_in[11];
  const float* W1_1  = (const float*)d_in[12];
  const float* b1_1  = (const float*)d_in[13];
  const float* W2_1  = (const float*)d_in[14];
  const float* b2_1  = (const float*)d_in[15];
  const float* Wvn   = (const float*)d_in[16];
  const float* bvn   = (const float*)d_in[17];
  const float* Wlin  = (const float*)d_in[18];
  const float* blin  = (const float*)d_in[19];
  float* out = (float*)d_out;

  // workspace carve-up
  float* h    = (float*)d_ws;             // 8388608
  float* agg  = h + HN;                   // 8388608
  float* vn   = agg + HN;                 // 131072
  float* s0   = vn + (long)NG * CD;       // 65536
  float* s1   = s0 + NN;                  // 65536
  float* wv   = s1 + NN;                  // 262144
  __bf16* cWenc = (__bf16*)(wv + NEDGE);  // 4096
  __bf16* cW1_0 = cWenc + IND * CD;       // 32768 each
  __bf16* cW2_0 = cW1_0 + CD * 2 * CD;
  __bf16* cW1_1 = cW2_0 + 2 * CD * CD;
  __bf16* cW2_1 = cW1_1 + CD * 2 * CD;

  // 1. weights -> bf16 column-major
  wconvert_kernel<<<(IND * CD + 255) / 256, 256, 0, stream>>>(Wenc, cWenc, IND, CD);
  wconvert_kernel<<<(CD * 2 * CD + 255) / 256, 256, 0, stream>>>(W1_0, cW1_0, CD, 2 * CD);
  wconvert_kernel<<<(2 * CD * CD + 255) / 256, 256, 0, stream>>>(W2_0, cW2_0, 2 * CD, CD);
  wconvert_kernel<<<(CD * 2 * CD + 255) / 256, 256, 0, stream>>>(W1_1, cW1_1, CD, 2 * CD);
  wconvert_kernel<<<(2 * CD * CD + 255) / 256, 256, 0, stream>>>(W2_1, cW2_1, 2 * CD, CD);

  // 2. encoder
  encode_kernel<<<NN / 64, 128, 0, stream>>>(x, cWenc, benc, h);

  // 3. layer 0: messages -> MLP (relu out) -> virtual node
  message_agg_kernel<<<NG, 256, 0, stream>>>(h, row, col, attr, tab0, agg);
  mlp_kernel<<<NN / 64, 128, 0, stream>>>(h, agg, cW1_0, b1_0, cW2_0, b2_0, h, 1);
  vnode_kernel<<<NG, CD, 0, stream>>>(h, Wvn, bvn, vn);

  // 4. layer 1: h += vn[batch]; messages; MLP (no relu)
  addvn_kernel<<<(HN / 4 + 255) / 256, 256, 0, stream>>>(h, vn);
  message_agg_kernel<<<NG, 256, 0, stream>>>(h, row, col, attr, tab1, agg);
  mlp_kernel<<<NN / 64, 128, 0, stream>>>(h, agg, cW1_1, b1_1, cW2_1, b2_1, h, 0);

  // 5. edge scoring
  score_kernel<<<(NN + 255) / 256, 256, 0, stream>>>(h, Wlin, s0, s1);
  edgew_kernel<<<(NEDGE + 255) / 256, 256, 0, stream>>>(row, col, s0, s1, blin, wv,
                                                        out + OFF_W);

  // 6. per-graph top-K split + index/attr/w outputs
  select_kernel<<<NG, EPG, 0, stream>>>(wv, row, col, attr, out);

  // 7. h and batch replicated into both tuples
  copyh_kernel<<<(HN / 4 + 255) / 256, 256, 0, stream>>>(h, out);
  batch_kernel<<<(NN + 255) / 256, 256, 0, stream>>>(out);
}
